// net_LSTM_21827023798541
// MI455X (gfx1250) — matrix operand order repaired and checked by reading, not compiled, not run on hardware
//
#include <hip/hip_runtime.h>
#include <hip/hip_bf16.h>

// LSTM: T=4096, B=256, I=128, H=128.
// Batch-parallel persistent workgroups: 16 WGs x 512 threads, each WG owns 16
// batch rows and runs the full 4096-step recurrence independently.
//
// Weights stay in VGPRs for the whole scan (128 VGPRs/lane of bf16 B-matrix
// WMMA fragments, loaded once fp32->bf16 from global). Waves 0-7 own the i,f
// gate tiles for h-columns [16w,16w+16); waves 8-15 own g,o. Per step only the
// double-buffered A panel (x_t ++ h_{t-1}, bf16) streams from LDS:
// 16 ds_load_b128 feed 16 v_wmma_f32_16x16x32_bf16 per wave.
//
// Critical-path layout per step:
//   GEMM (all waves)
//   pre-barrier : waves 0-7 compute sigmoid(i),sigmoid(f) in-place;
//                 waves 8-15 compute tanh(g),sigmoid(o), hand them off via a
//                 packed float2 LDS scratch, then stage x_{t+1} into the other
//                 A panel and prefetch x_{t+2} (global_prefetch_b8).
//   barrier
//   post-barrier: waves 0-7 do c = f*c + i*g, h = o*tanh(c) (fp32 in regs),
//                 write h to global out and recycle bf16 h into the other panel.
//   barrier

#define T_STEPS 4096
#define B_SZ    256
#define I_SZ    128
#define H_SZ    128
#define ROWS_WG 16               // batch rows per workgroup
#define NUM_WG  (B_SZ / ROWS_WG) // 16
#define A_STRIDE 264             // padded row stride (bf16 elems) for A panel
#define JPAD     130             // padded column stride (float2) for gate scratch

typedef __attribute__((ext_vector_type(16))) __bf16 v16bf;
typedef __attribute__((ext_vector_type(8)))  float  v8f;

__device__ __forceinline__ __bf16 f2bf(float f) {
    union { float f; unsigned u; } a; a.f = f;
    unsigned r = a.u + 0x7FFFu + ((a.u >> 16) & 1u);   // RNE
    union { unsigned short s; __bf16 b; } o; o.s = (unsigned short)(r >> 16);
    return o.b;
}

__device__ __forceinline__ float fast_sigmoid(float x) {
    return 1.0f / (1.0f + __expf(-x));
}
__device__ __forceinline__ float fast_tanh(float x) {
    return 1.0f - 2.0f / (__expf(2.0f * x) + 1.0f);
}

__device__ __forceinline__ v8f splat8(float v) {
    v8f r;
#pragma unroll
    for (int i = 0; i < 8; ++i) r[i] = v;
    return r;
}

union Frag { uint4 q[2]; v16bf v; };

// Convert 16 consecutive fp32 into one bf16 WMMA B fragment.
__device__ __forceinline__ Frag load_bfrag(const float* __restrict__ src) {
    Frag f;
#pragma unroll
    for (int c = 0; c < 2; ++c) {
        float4 v0 = ((const float4*)src)[c * 2 + 0];
        float4 v1 = ((const float4*)src)[c * 2 + 1];
        union { __bf16 b[8]; uint4 q; } p;
        p.b[0] = f2bf(v0.x); p.b[1] = f2bf(v0.y);
        p.b[2] = f2bf(v0.z); p.b[3] = f2bf(v0.w);
        p.b[4] = f2bf(v1.x); p.b[5] = f2bf(v1.y);
        p.b[6] = f2bf(v1.z); p.b[7] = f2bf(v1.w);
        f.q[c] = p.q;
    }
    return f;
}

__device__ __forceinline__ void stage_x(const float* __restrict__ xrow,
                                        __bf16* __restrict__ dst) {
    float4 xa = ((const float4*)xrow)[0];
    float4 xb = ((const float4*)xrow)[1];
    union { __bf16 b[8]; uint4 q; } pk;
    pk.b[0] = f2bf(xa.x); pk.b[1] = f2bf(xa.y);
    pk.b[2] = f2bf(xa.z); pk.b[3] = f2bf(xa.w);
    pk.b[4] = f2bf(xb.x); pk.b[5] = f2bf(xb.y);
    pk.b[6] = f2bf(xb.z); pk.b[7] = f2bf(xb.w);
    *(uint4*)dst = pk.q;
}

__global__ __launch_bounds__(512, 1)
void lstm_fused_kernel(const float* __restrict__ x,
                       const float* __restrict__ W_ih,
                       const float* __restrict__ W_hh,
                       const float* __restrict__ b_ih,
                       const float* __restrict__ b_hh,
                       float* __restrict__ out)   // [T,B,H] ++ [B,H] ++ [B,H]
{
    __shared__ __align__(16) __bf16 Alds[2][ROWS_WG][A_STRIDE]; // double buffer
    __shared__ __align__(8)  float  Gscr[ROWS_WG][JPAD][2];     // {tanh g, sig o}

    const int tid   = threadIdx.x;
    const int wave  = tid >> 5;         // 0..15
    const int lane  = tid & 31;
    const int gpair = wave >> 3;        // 0: gates i,f   1: gates g,o
    const int wslot = wave & 7;         // h-column group
    const int r0    = blockIdx.x * ROWS_WG;

    const int jl   = lane & 15;
    const int hi   = lane >> 4;
    const int jcol = wslot * 16 + jl;   // h column 0..127 owned by this lane

    const int g0 = gpair * 2;           // torch gate order i,f,g,o
    const int g1 = g0 + 1;

    // ---- persistent B fragments: W row n = g*128 + jcol, contiguous K ----
    const int nrow0 = g0 * H_SZ + jcol;
    const int nrow1 = g1 * H_SZ + jcol;
    Frag Bk[8][2];
#pragma unroll
    for (int kk = 0; kk < 8; ++kk) {
        const int kb = kk * 32 + hi * 16;    // 16-elem chunk, never straddles I
        const float* s0 = (kb < I_SZ) ? (W_ih + (size_t)nrow0 * I_SZ + kb)
                                      : (W_hh + (size_t)nrow0 * H_SZ + (kb - I_SZ));
        const float* s1 = (kb < I_SZ) ? (W_ih + (size_t)nrow1 * I_SZ + kb)
                                      : (W_hh + (size_t)nrow1 * H_SZ + (kb - I_SZ));
        Bk[kk][0] = load_bfrag(s0);
        Bk[kk][1] = load_bfrag(s1);
    }

    const float bias0 = b_ih[g0 * H_SZ + jcol] + b_hh[g0 * H_SZ + jcol];
    const float bias1 = b_ih[g1 * H_SZ + jcol] + b_hh[g1 * H_SZ + jcol];

    // x staging coordinates for waves 8..15 (256 threads: 16 rows x 16 chunks of 8).
    const int stid = tid - 256;
    const int sm   = stid >> 4;
    const int sk   = (stid & 15) << 3;

    float hreg[8], creg[8];
#pragma unroll
    for (int r = 0; r < 8; ++r) { hreg[r] = 0.0f; creg[r] = 0.0f; }

    // ---- prologue: init panel 0 with (x_0, h_{-1}=0) ----
    if (gpair == 0) {
#pragma unroll
        for (int r = 0; r < 8; ++r)
            Alds[0][r + 8 * hi][I_SZ + jcol] = f2bf(0.0f);
    } else {
        stage_x(x + ((size_t)0 * B_SZ + r0 + sm) * I_SZ + sk, &Alds[0][sm][sk]);
    }
    __syncthreads();

    const size_t out_step = (size_t)B_SZ * H_SZ;

    for (int t = 0; t < T_STEPS; ++t) {
        const int p = t & 1;
        const __bf16* Ap = &Alds[p][lane & 15][0];

        // ---- gate GEMM: B resident in VGPRs, only A streamed from LDS ----
        v8f acc0 = splat8(bias0);
        v8f acc1 = splat8(bias1);
#pragma unroll
        for (int kk = 0; kk < 8; ++kk) {
            Frag a;
            a.q[0] = *(const uint4*)(Ap + kk * 32 + hi * 16);
            a.q[1] = *(const uint4*)(Ap + kk * 32 + hi * 16 + 8);
            acc0 = __builtin_amdgcn_wmma_f32_16x16x32_bf16(
                false, a.v, false, Bk[kk][0].v, (short)0, acc0, false, false);
            acc1 = __builtin_amdgcn_wmma_f32_16x16x32_bf16(
                false, a.v, false, Bk[kk][1].v, (short)0, acc1, false, false);
        }

        // ---- pre-barrier: both wave groups activate their gates in parallel ----
        if (gpair == 1) {
#pragma unroll
            for (int r = 0; r < 8; ++r) {
                const int m = r + 8 * hi;
                float2 go;
                go.x = fast_tanh(acc0[r]);      // tanh(g)
                go.y = fast_sigmoid(acc1[r]);   // sigmoid(o)
                *(float2*)&Gscr[m][jcol][0] = go;
            }
            // stage x_{t+1} into the other panel (its readers are long done)
            if (t + 1 < T_STEPS) {
                stage_x(x + ((size_t)(t + 1) * B_SZ + r0 + sm) * I_SZ + sk,
                        &Alds[p ^ 1][sm][sk]);
                if (t + 2 < T_STEPS) {
                    __builtin_prefetch(
                        x + ((size_t)(t + 2) * B_SZ + r0 + sm) * I_SZ + sk, 0, 1);
                }
            }
        } else {
#pragma unroll
            for (int r = 0; r < 8; ++r) {
                acc0[r] = fast_sigmoid(acc0[r]);   // i
                acc1[r] = fast_sigmoid(acc1[r]);   // f
            }
        }
        __syncthreads();   // activated g,o published; GEMM reads of panel p done

        // ---- post-barrier: minimal cell update on waves 0-7 ----
        if (gpair == 0) {
            float* out_t = out + (size_t)t * out_step;
#pragma unroll
            for (int r = 0; r < 8; ++r) {
                const int m = r + 8 * hi;
                const float2 go = *(const float2*)&Gscr[m][jcol][0];
                creg[r] = acc1[r] * creg[r] + acc0[r] * go.x;
                hreg[r] = go.y * fast_tanh(creg[r]);
                out_t[(size_t)(r0 + m) * H_SZ + jcol] = hreg[r];
                Alds[p ^ 1][m][I_SZ + jcol] = f2bf(hreg[r]);  // h for t+1
            }
        }
        __syncthreads();   // panel p^1 (x_{t+1}, h_t) complete
    }

    // ---- final h_n, c_n ----
    if (gpair == 0) {
        float* hout = out + (size_t)T_STEPS * out_step;
        float* cout = hout + out_step;
#pragma unroll
        for (int r = 0; r < 8; ++r) {
            const int m = r + 8 * hi;
            hout[(size_t)(r0 + m) * H_SZ + jcol] = hreg[r];
            cout[(size_t)(r0 + m) * H_SZ + jcol] = creg[r];
        }
    }
}

extern "C" void kernel_launch(void* const* d_in, const int* in_sizes, int n_in,
                              void* d_out, int out_size, void* d_ws, size_t ws_size,
                              hipStream_t stream) {
    (void)in_sizes; (void)n_in; (void)out_size; (void)d_ws; (void)ws_size;
    const float* x    = (const float*)d_in[0];
    const float* W_ih = (const float*)d_in[1];
    const float* W_hh = (const float*)d_in[2];
    const float* b_ih = (const float*)d_in[3];
    const float* b_hh = (const float*)d_in[4];
    float* out = (float*)d_out;

    lstm_fused_kernel<<<NUM_WG, 512, 0, stream>>>(x, W_ih, W_hh, b_ih, b_hh, out);
}
